// GTLayer_55800215110124
// MI455X (gfx1250) — compile-verified, matrix operand order checked by reading, not run on hardware
//
#include <hip/hip_runtime.h>

#define B_ 2
#define N_ 4096
#define D_ 256
#define H_ 8
#define HD_ 32
#define NEG_SLOPE 0.01f
#define LN_EPS 1e-5f

typedef __attribute__((ext_vector_type(16))) __bf16 v16bf;
typedef __attribute__((ext_vector_type(8)))  float  v8f;
typedef __attribute__((ext_vector_type(4)))  float  f4;
typedef __attribute__((ext_vector_type(4)))  int    v4i;

__device__ __forceinline__ unsigned short f2bfbits(float x) {
  unsigned u = __builtin_bit_cast(unsigned, x);
  u += 0x7fffu + ((u >> 16) & 1u);          // round-to-nearest-even
  return (unsigned short)(u >> 16);
}

struct Frag32B { uint4 lo; uint4 hi; };
__device__ __forceinline__ v16bf mkfrag(const void* plo, const void* phi) {
  Frag32B f;
  f.lo = *(const uint4*)plo;
  f.hi = *(const uint4*)phi;
  return __builtin_bit_cast(v16bf, f);
}

// ---- CDNA5 async global->LDS (ASYNCcnt path), guarded for portability ----
#if defined(__AMDGCN__) && __has_builtin(__builtin_amdgcn_global_load_async_to_lds_b128)
#define HAVE_ASYNC_LDS 1
#define AS1 __attribute__((address_space(1)))
#define AS3 __attribute__((address_space(3)))
__device__ __forceinline__ void async_cp16(void* lds, const void* g) {
  // signature (probe-confirmed by diagnostic): (AS1 v4i*, AS3 v4i*, imm, imm)
  __builtin_amdgcn_global_load_async_to_lds_b128(
      (AS1 v4i*)(AS1 const void*)g, (AS3 v4i*)lds, 0, 0);
}
#if __has_builtin(__builtin_amdgcn_s_wait_asynccnt)
#define ASYNC_WAIT(n) __builtin_amdgcn_s_wait_asynccnt(n)
#else
#define ASYNC_WAIT(n) asm volatile("s_wait_asynccnt %0" ::"i"(n) : "memory")
#endif
#else
#define HAVE_ASYNC_LDS 0
__device__ __forceinline__ void async_cp16(void* lds, const void* g) {
  *(uint4*)lds = *(const uint4*)g;          // fallback: through VGPRs
}
#define ASYNC_WAIT(n) ((void)0)
#endif

// ---------------------------------------------------------------------------
// Stage 0a: h (f32) -> hbf (bf16), once.  4 elems / thread, packed b64 stores.
// ---------------------------------------------------------------------------
__launch_bounds__(256)
__global__ void k_cvt(const float* __restrict__ src, unsigned short* __restrict__ dst) {
  const int i = blockIdx.x * 256 + threadIdx.x;     // group of 4 floats
  f4 v = reinterpret_cast<const f4*>(src)[i];
  unsigned lo = (unsigned)f2bfbits(v.x) | ((unsigned)f2bfbits(v.y) << 16);
  unsigned hi = (unsigned)f2bfbits(v.z) | ((unsigned)f2bfbits(v.w) << 16);
  uint2 p; p.x = lo; p.y = hi;
  reinterpret_cast<uint2*>(dst)[i] = p;
}

// ---------------------------------------------------------------------------
// Stage 0b: W_l, W_r (f32, [k][c]) -> bf16 transposed [c][k] so that
// B-matrix fragments are contiguous 16B runs in LDS.
// ---------------------------------------------------------------------------
__launch_bounds__(256)
__global__ void k_cvtw(const float* __restrict__ Wl, const float* __restrict__ Wr,
                       unsigned short* __restrict__ wlt, unsigned short* __restrict__ wrt) {
  const int idx = blockIdx.x * 256 + threadIdx.x;   // [0, D*D)
  const int k = idx >> 8, c = idx & 255;
  wlt[c * D_ + k] = f2bfbits(Wl[idx]);
  wrt[c * D_ + k] = f2bfbits(Wr[idx]);
}

// ---------------------------------------------------------------------------
// Stage 1: fl = h@W_l, fr = h@W_r via bf16 WMMA.
// K-loop: double-buffered async global->LDS copies (no VGPR staging, no
// conversion) -> ds_load_b128 fragments -> v_wmma_f32_16x16x32_bf16.
// Grid: (B*N/128, D/64), block 256 (8 waves).
// Dynamic LDS: stage = [2][128][40] A-bf16 + [2][2][64][40] B-bf16 = 40960B,
// aliased after the K-loop by the fp32 out tile [128][68] (34816B).
// ---------------------------------------------------------------------------
__launch_bounds__(256)
__global__ void k_proj(const unsigned short* __restrict__ hbf,
                       const unsigned short* __restrict__ wlt,
                       const unsigned short* __restrict__ wrt,
                       const float* __restrict__ a_l,
                       const float* __restrict__ a_r,
                       float* __restrict__ frw,
                       float* __restrict__ slw,
                       float* __restrict__ srw) {
  extern __shared__ char smem[];
  __bf16* Ast = (__bf16*)smem;                       // [2][128][40]
  __bf16* Bst = (__bf16*)(smem + 2 * 128 * 40 * 2);  // [2][2][64][40]
  float*  Oub = (float*)smem;                        // [128][68] (reuse)

  const int t    = threadIdx.x;
  const int lane = t & 31;
  const int wave = t >> 5;
  const int row0 = blockIdx.x * 128;
  const int col0 = blockIdx.y * 64;
  const int b    = row0 >> 12;

  // issue one 32-wide K-tile of A (8KB) + both B tiles (8KB): 4 b128/thread
  auto issue_tile = [&](int buf, int kt) {
    const int k0 = kt * 32;
#pragma unroll
    for (int s = 0; s < 2; ++s) {                    // A: 512 chunks of 16B
      const int chunk = t + s * 256;
      const int row = chunk >> 2, q = chunk & 3;
      const unsigned short* src = hbf + (size_t)(row0 + row) * D_ + k0 + q * 8;
      async_cp16(Ast + ((size_t)(buf * 128 + row)) * 40 + q * 8, src);
    }
#pragma unroll
    for (int s = 0; s < 2; ++s) {                    // B: 2 mats x 64 cols x 4
      const int chunk = t + s * 256;
      const int m = chunk >> 8, c = (chunk >> 2) & 63, q = chunk & 3;
      const unsigned short* w = m ? wrt : wlt;
      const unsigned short* src = w + (size_t)(col0 + c) * D_ + k0 + q * 8;
      async_cp16(Bst + ((size_t)((buf * 2 + m) * 64 + c)) * 40 + q * 8, src);
    }
  };

  v8f zero = {};
  v8f accL[4], accR[4];
#pragma unroll
  for (int ct = 0; ct < 4; ++ct) { accL[ct] = zero; accR[ct] = zero; }

  issue_tile(0, 0);
  for (int kt = 0; kt < 8; ++kt) {
    const int buf = kt & 1;
    if (kt < 7) {
      issue_tile(buf ^ 1, kt + 1);
      ASYNC_WAIT(4);                 // newest 4/thread = next tile; current done
    } else {
      ASYNC_WAIT(0);
    }
    __syncthreads();                 // current tile visible to all waves

    const int arow = wave * 16 + (lane & 15);
    const int kb   = (lane >> 4) * 8;
    const __bf16* Ab = Ast + ((size_t)(buf * 128 + arow)) * 40;
    v16bf a = mkfrag(Ab + kb, Ab + kb + 16);
#pragma unroll
    for (int ct = 0; ct < 4; ++ct) {
      const int bcol = ct * 16 + (lane & 15);
      const __bf16* Bl = Bst + ((size_t)((buf * 2 + 0) * 64 + bcol)) * 40;
      const __bf16* Br = Bst + ((size_t)((buf * 2 + 1) * 64 + bcol)) * 40;
      v16bf bl = mkfrag(Bl + kb, Bl + kb + 16);
      v16bf br = mkfrag(Br + kb, Br + kb + 16);
      accL[ct] = __builtin_amdgcn_wmma_f32_16x16x32_bf16(
          false, a, false, bl, (short)0, accL[ct], false, false);
      accR[ct] = __builtin_amdgcn_wmma_f32_16x16x32_bf16(
          false, a, false, br, (short)0, accR[ct], false, false);
    }
    __syncthreads();                 // protect buffer reuse (tile kt+2)
  }

  // ---- fl tiles -> LDS, reduce sl = <leaky(fl_head), a_l> ----
#pragma unroll
  for (int ct = 0; ct < 4; ++ct) {
    const int colb = ct * 16 + (lane & 15);
#pragma unroll
    for (int r = 0; r < 8; ++r) {
      const int rowi = wave * 16 + r + ((lane >= 16) ? 8 : 0);
      Oub[rowi * 68 + colb] = accL[ct][r];
    }
  }
  __syncthreads();
  {
    const int r  = t & 127;
    const int hh = t >> 7;
    float s = 0.f;
#pragma unroll
    for (int kk = 0; kk < HD_; ++kk) {
      float v = Oub[r * 68 + hh * HD_ + kk];
      v = (v >= 0.f) ? v : NEG_SLOPE * v;
      s += v * a_l[kk];
    }
    const int head = (col0 >> 5) + hh;
    const int n    = (row0 + r) & (N_ - 1);
    slw[(size_t)(b * H_ + head) * N_ + n] = s;
  }
  __syncthreads();

  // ---- fr tiles -> LDS; reduce sr; stream fr tile to HBM ----
#pragma unroll
  for (int ct = 0; ct < 4; ++ct) {
    const int colb = ct * 16 + (lane & 15);
#pragma unroll
    for (int r = 0; r < 8; ++r) {
      const int rowi = wave * 16 + r + ((lane >= 16) ? 8 : 0);
      Oub[rowi * 68 + colb] = accR[ct][r];
    }
  }
  __syncthreads();
  {
    const int r  = t & 127;
    const int hh = t >> 7;
    float s = 0.f;
#pragma unroll
    for (int kk = 0; kk < HD_; ++kk) {
      float v = Oub[r * 68 + hh * HD_ + kk];
      v = (v >= 0.f) ? v : NEG_SLOPE * v;
      s += v * a_r[kk];
    }
    const int head = (col0 >> 5) + hh;
    const int n    = (row0 + r) & (N_ - 1);
    srw[(size_t)(b * H_ + head) * N_ + n] = s;
  }
  for (int idx = t; idx < 128 * 16; idx += 256) {
    const int r  = idx >> 4;
    const int c4 = (idx & 15) * 4;
    f4 v = *reinterpret_cast<const f4*>(&Oub[r * 68 + c4]);
    *reinterpret_cast<f4*>(frw + (size_t)(row0 + r) * D_ + col0 + c4) = v;
  }
}

// ---------------------------------------------------------------------------
// Stage 2: per (b,h): softmax over sr (row-independent!) and
// c[b,h,:] = sum_j softmax(sr)_j * fr[b,h,j,:].  Grid: B*H blocks.
// ---------------------------------------------------------------------------
__launch_bounds__(256)
__global__ void k_scores(const float* __restrict__ srw,
                         const float* __restrict__ frw,
                         float* __restrict__ cvec) {
  __shared__ float red[256];
  __shared__ float red2[8][32];
  __shared__ float s_mx, s_sum;
  const int t  = threadIdx.x;
  const int b  = blockIdx.x / H_;
  const int hd = blockIdx.x % H_;
  const float* s = srw + (size_t)(b * H_ + hd) * N_;

  float mx = -3.4e38f;
  for (int j = t; j < N_; j += 256) mx = fmaxf(mx, s[j]);
  red[t] = mx; __syncthreads();
  for (int o = 128; o > 0; o >>= 1) {
    if (t < o) red[t] = fmaxf(red[t], red[t + o]);
    __syncthreads();
  }
  if (t == 0) s_mx = red[0];
  __syncthreads();
  mx = s_mx;

  float sum = 0.f;
  for (int j = t; j < N_; j += 256) sum += __expf(s[j] - mx);
  red[t] = sum; __syncthreads();
  for (int o = 128; o > 0; o >>= 1) {
    if (t < o) red[t] += red[t + o];
    __syncthreads();
  }
  if (t == 0) s_sum = red[0];
  __syncthreads();
  const float inv = 1.f / s_sum;

  const int d = t & 31, g = t >> 5;
  float acc = 0.f;
  for (int j = g * (N_ / 8); j < (g + 1) * (N_ / 8); ++j) {
    acc += __expf(s[j] - mx) * frw[((size_t)b * N_ + j) * D_ + hd * HD_ + d];
  }
  red2[g][d] = acc; __syncthreads();
  if (g == 0) {
    float tot = 0.f;
#pragma unroll
    for (int gg = 0; gg < 8; ++gg) tot += red2[gg][d];
    cvec[b * D_ + hd * HD_ + d] = tot * inv;
  }
}

// ---------------------------------------------------------------------------
// Stage 3: head means: msl[b,i], msr[b,j].
// ---------------------------------------------------------------------------
__launch_bounds__(256)
__global__ void k_mean(const float* __restrict__ slw,
                       const float* __restrict__ srw,
                       float* __restrict__ msl, float* __restrict__ msr) {
  const int idx = blockIdx.x * 256 + threadIdx.x;
  const int b = idx >> 12, n = idx & (N_ - 1);
  float al = 0.f, ar = 0.f;
#pragma unroll
  for (int hh = 0; hh < H_; ++hh) {
    al += slw[(size_t)(b * H_ + hh) * N_ + n];
    ar += srw[(size_t)(b * H_ + hh) * N_ + n];
  }
  msl[idx] = al * (1.f / H_);
  msr[idx] = ar * (1.f / H_);
}

// ---------------------------------------------------------------------------
// Stage 4: collapsed final GEMV: w[b] = v[b] @ W_final (context is rank-1).
// ---------------------------------------------------------------------------
__launch_bounds__(256)
__global__ void k_final(const float* __restrict__ cvec,
                        const float* __restrict__ Wf,
                        float* __restrict__ wvec) {
  const int b = blockIdx.x, o = threadIdx.x;
  float acc = 0.f;
  for (int d = 0; d < D_; ++d) acc += cvec[b * D_ + d] * Wf[(size_t)d * D_ + o];
  wvec[b * D_ + o] = acc;
}

// ---------------------------------------------------------------------------
// Stage 5: LayerNorm(h + w[b]).  Grid: B*N blocks x 256.
// ---------------------------------------------------------------------------
__launch_bounds__(256)
__global__ void k_ln(const float* __restrict__ h, const float* __restrict__ wvec,
                     const float* __restrict__ gam, const float* __restrict__ bet,
                     float* __restrict__ out) {
  __shared__ float red[256];
  __shared__ float s_mu, s_rstd;
  const int t = threadIdx.x;
  const int gr = blockIdx.x;
  const int b = gr >> 12;
  const float x = h[(size_t)gr * D_ + t] + wvec[b * D_ + t];
  red[t] = x; __syncthreads();
  for (int o = 128; o > 0; o >>= 1) {
    if (t < o) red[t] += red[t + o];
    __syncthreads();
  }
  if (t == 0) s_mu = red[0] * (1.f / D_);
  __syncthreads();
  const float xc = x - s_mu;
  red[t] = xc * xc; __syncthreads();
  for (int o = 128; o > 0; o >>= 1) {
    if (t < o) red[t] += red[t + o];
    __syncthreads();
  }
  if (t == 0) s_rstd = rsqrtf(red[0] * (1.f / D_) + LN_EPS);
  __syncthreads();
  out[(size_t)gr * D_ + t] = xc * s_rstd * gam[t] + bet[t];
}

// ---------------------------------------------------------------------------
// Stage 6: attn[b,i,j] = msl[b,i] + msr[b,j] — 134MB stream, NT b128 stores.
// ---------------------------------------------------------------------------
__launch_bounds__(256)
__global__ void k_attn(const float* __restrict__ msl,
                       const float* __restrict__ msr,
                       float* __restrict__ attn) {
  const int t = threadIdx.x;
  const int gr = blockIdx.x;
  const int b = gr >> 12;
  const float si = msl[gr];
  const f4* m4 = reinterpret_cast<const f4*>(msr + b * N_);
  f4* o4 = reinterpret_cast<f4*>(attn + (size_t)gr * N_);
  for (int c = t; c < N_ / 4; c += 256) {
    f4 v = m4[c];
    f4 r = { si + v.x, si + v.y, si + v.z, si + v.w };
    __builtin_nontemporal_store(r, o4 + c);
  }
}

// ---------------------------------------------------------------------------
extern "C" void kernel_launch(void* const* d_in, const int* in_sizes, int n_in,
                              void* d_out, int out_size, void* d_ws, size_t ws_size,
                              hipStream_t stream) {
  (void)in_sizes; (void)n_in; (void)out_size; (void)ws_size;
  const float* h  = (const float*)d_in[0];
  const float* Wl = (const float*)d_in[1];
  const float* Wr = (const float*)d_in[2];
  const float* al = (const float*)d_in[3];
  const float* ar = (const float*)d_in[4];
  const float* Wf = (const float*)d_in[5];
  const float* g  = (const float*)d_in[6];
  const float* be = (const float*)d_in[7];

  float* out  = (float*)d_out;
  float* hout = out;
  float* attn = out + (size_t)B_ * N_ * D_;

  float* ws   = (float*)d_ws;
  float* frw  = ws; ws += (size_t)B_ * N_ * D_;
  float* slw  = ws; ws += (size_t)B_ * H_ * N_;
  float* srw  = ws; ws += (size_t)B_ * H_ * N_;
  float* msl  = ws; ws += (size_t)B_ * N_;
  float* msr  = ws; ws += (size_t)B_ * N_;
  float* cvec = ws; ws += (size_t)B_ * D_;
  float* wvec = ws; ws += (size_t)B_ * D_;
  unsigned short* hbf = (unsigned short*)ws;
  unsigned short* wlt = hbf + (size_t)B_ * N_ * D_;
  unsigned short* wrt = wlt + (size_t)D_ * D_;

  const int smem_proj = 2 * 128 * 40 * 2 + 2 * 2 * 64 * 40 * 2;  // 40960B

  hipLaunchKernelGGL(k_cvt, dim3((B_ * N_ * D_) / 1024), dim3(256), 0, stream, h, hbf);
  hipLaunchKernelGGL(k_cvtw, dim3((D_ * D_) / 256), dim3(256), 0, stream, Wl, Wr, wlt, wrt);
  hipLaunchKernelGGL(k_proj, dim3((B_ * N_) / 128, D_ / 64), dim3(256), smem_proj, stream,
                     hbf, wlt, wrt, al, ar, frw, slw, srw);
  hipLaunchKernelGGL(k_scores, dim3(B_ * H_), dim3(256), 0, stream, srw, frw, cvec);
  hipLaunchKernelGGL(k_mean, dim3((B_ * N_) / 256), dim3(256), 0, stream,
                     slw, srw, msl, msr);
  hipLaunchKernelGGL(k_final, dim3(B_), dim3(256), 0, stream, cvec, Wf, wvec);
  hipLaunchKernelGGL(k_ln, dim3(B_ * N_), dim3(256), 0, stream, h, wvec, g, be, hout);
  hipLaunchKernelGGL(k_attn, dim3(B_ * N_), dim3(256), 0, stream, msl, msr, attn);
}
